// GraphHallucinationRCNN_37409165148198
// MI455X (gfx1250) — matrix operand order repaired and checked by reading, not compiled, not run on hardware
//
#include <hip/hip_runtime.h>

#define N_NODES 768
#define C_FEAT  1024
#define HID     256
#define HEADS   4
#define TOPK    5
#define SLOTS   6                 // TOPK + self loop
#define NEDGES  (N_NODES*SLOTS)
#define XK      1028              // C + 4 geom
#define XKP     1056              // padded to multiple of 32
#define OUTC    775               // 768 rel + 2 logits + 4 boxes + 1 prob
#define NEG_SLOPE 0.2f

// async global->LDS (gfx1250): guarded so host pass / older toolchains fall back
#if defined(__AMDGCN__) && __has_builtin(__builtin_amdgcn_global_load_async_to_lds_b128) && __has_builtin(__builtin_amdgcn_s_wait_asynccnt)
#define ASYNC_LDS 1
// builtin expects: (int __vector(4) AS1 *src, int __vector(4) AS3 *dst, imm offset, imm cpol)
typedef int v4i_vs __attribute__((vector_size(16)));
typedef __attribute__((address_space(1))) v4i_vs* g_v4i_p;
typedef __attribute__((address_space(3))) v4i_vs* l_v4i_p;
#else
#define ASYNC_LDS 0
#endif

typedef __attribute__((ext_vector_type(16))) __bf16 bf16x16;
typedef __attribute__((ext_vector_type(8)))  float  f32x8;

// ---------- helpers ----------
__device__ __forceinline__ __bf16 f2bf(float x) {
  union { float f; unsigned u; } v; v.f = x;
  unsigned r = v.u + 0x7FFFu + ((v.u >> 16) & 1u);   // round-to-nearest-even
  unsigned short hs = (unsigned short)(r >> 16);
  __bf16 o; __builtin_memcpy(&o, &hs, 2); return o;
}
// monotonic uint encoding of float for atomicMax-based segment max
__device__ __forceinline__ unsigned fkey(float f) {
  union { float f; unsigned u; } v; v.f = f;
  return (v.u & 0x80000000u) ? ~v.u : (v.u | 0x80000000u);
}
__device__ __forceinline__ float funkey(unsigned u) {
  union { float f; unsigned u; } v;
  v.u = (u & 0x80000000u) ? (u ^ 0x80000000u) : ~u;
  return v.f;
}

__global__ void k_zero(float* p, int n) {
  int i = blockIdx.x*blockDim.x + threadIdx.x;
  if (i < n) p[i] = 0.f;
}

// ---------- stage 0: geom + bf16 staging of x = [feats | geom | 0-pad] ----------
__global__ void k_build_x(const float* __restrict__ feats, const float* __restrict__ boxes,
                          const int* __restrict__ ih, const int* __restrict__ iw,
                          float* __restrict__ geom, __bf16* __restrict__ xbf) {
  int t = blockIdx.x*blockDim.x + threadIdx.x;
  int n = t / XKP, c = t % XKP;
  if (n >= N_NODES) return;
  float w = (float)iw[0], h = (float)ih[0];
  if (c == 0) {
    float x0 = boxes[n*4+0]/w, y0 = boxes[n*4+1]/h;
    float x1 = boxes[n*4+2]/w, y1 = boxes[n*4+3]/h;
    geom[n*4+0]=x0; geom[n*4+1]=y0; geom[n*4+2]=x1-x0; geom[n*4+3]=y1-y0;
  }
  float v;
  if (c < C_FEAT) v = feats[n*C_FEAT + c];
  else if (c < XK) {
    float x0 = boxes[n*4+0]/w, y0 = boxes[n*4+1]/h;
    float x1 = boxes[n*4+2]/w, y1 = boxes[n*4+3]/h;
    float g4[4] = {x0, y0, x1-x0, y1-y0};
    v = g4[c - C_FEAT];
  } else v = 0.f;
  xbf[n*XKP + c] = f2bf(v);
}

// cast fp32 weights to bf16, zero-padding extra rows (same #cols)
__global__ void k_castpad(const float* __restrict__ src, __bf16* __restrict__ dst,
                          int srcRows, int cols, int dstRows) {
  int t = blockIdx.x*blockDim.x + threadIdx.x;
  if (t >= dstRows*cols) return;
  int r = t / cols;
  dst[t] = f2bf(r < srcRows ? src[t] : 0.f);
}

// ---------- WMMA bf16 GEMM: one wave -> 16x64 tile (A fragment reused 4x) ----------
__global__ __launch_bounds__(32)
void k_wmma_gemm(const __bf16* __restrict__ A, const __bf16* __restrict__ B,
                 float* __restrict__ D, int K, int lda, int ldb, int ldd, int transD) {
  int lane = threadIdx.x;
  int m0 = blockIdx.x * 16;
  int n0 = blockIdx.y * 64;
  int h  = lane >> 4;                       // K-half selector for A
  const __bf16* arow = A + (size_t)(m0 + (lane & 15)) * lda + h * 8;
  f32x8 acc0 = {}, acc1 = {}, acc2 = {}, acc3 = {};
  for (int k0 = 0; k0 < K; k0 += 32) {
    // A 16x32: comps0-7 = K[h*8..], comps8-15 = K[16+h*8..]
    bf16x16 a;
    const __bf16* pa = arow + k0;
    #pragma unroll
    for (int t = 0; t < 8; ++t) { a[t] = pa[t]; a[t+8] = pa[t+16]; }
    // B 32x16 x4: lane&15 = N col, comps0-7 = K[h*8..], comps8-15 = K[16+h*8..]
    const __bf16* pb = B + (size_t)k0 * ldb + n0 + (lane & 15);
    bf16x16 b0, b1, b2, b3;
    #pragma unroll
    for (int t = 0; t < 16; ++t) {
      int kk = (t < 8) ? (h*8 + t) : (16 + h*8 + (t-8));
      const __bf16* q = pb + (size_t)kk * ldb;
      b0[t] = q[0]; b1[t] = q[16]; b2[t] = q[32]; b3[t] = q[48];
    }
    __builtin_prefetch(pa + 32, 0, 0);      // next K tile -> global_prefetch_b8
    __builtin_prefetch(pb + 32*ldb, 0, 0);
    acc0 = __builtin_amdgcn_wmma_f32_16x16x32_bf16(false, a, false, b0, (short)0, acc0, false, false);
    acc1 = __builtin_amdgcn_wmma_f32_16x16x32_bf16(false, a, false, b1, (short)0, acc1, false, false);
    acc2 = __builtin_amdgcn_wmma_f32_16x16x32_bf16(false, a, false, b2, (short)0, acc2, false, false);
    acc3 = __builtin_amdgcn_wmma_f32_16x16x32_bf16(false, a, false, b3, (short)0, acc3, false, false);
  }
  // D: VGPR r -> M = r + 8*h, N = lane&15 (per 16-wide subtile)
  int nl = lane & 15;
  #pragma unroll
  for (int r = 0; r < 8; ++r) {
    int m = m0 + r + 8*h;
    float vals[4] = {acc0[r], acc1[r], acc2[r], acc3[r]};
    #pragma unroll
    for (int s = 0; s < 4; ++s) {
      int n = n0 + 16*s + nl;
      if (transD) D[(size_t)n*ldd + m] = vals[s];
      else        D[(size_t)m*ldd + n] = vals[s];
    }
  }
}

// ---------- fused rel[i,j] with relu inside the 256-length reduction ----------
__global__ __launch_bounds__(256)
void k_rel(const float* __restrict__ fa, const float* __restrict__ fbT,
           const float* __restrict__ boxes, const float* __restrict__ W1g,
           const float* __restrict__ b1, const float* __restrict__ W2r,
           const float* __restrict__ b2r, float* __restrict__ out) {
  __shared__ float sA[HID], sW2[HID], sG[4*HID];
  int i = blockIdx.y;
  int j = blockIdx.x * 256 + threadIdx.x;
#if ASYNC_LDS
  {
    int t = threadIdx.x;
    // sG: 1024 floats = 256 b128 async transfers; sW2: 64 b128 async transfers
    __builtin_amdgcn_global_load_async_to_lds_b128(
        (g_v4i_p)(const_cast<float*>(W1g) + t*4),
        (l_v4i_p)(&sG[t*4]), 0, 0);
    if (t < 64)
      __builtin_amdgcn_global_load_async_to_lds_b128(
          (g_v4i_p)(const_cast<float*>(W2r) + t*4),
          (l_v4i_p)(&sW2[t*4]), 0, 0);
    __builtin_amdgcn_s_wait_asynccnt(0);
  }
  sA[threadIdx.x] = fa[i*HID + threadIdx.x] + b1[threadIdx.x];
#else
  for (int t = threadIdx.x; t < HID; t += 256) {
    sA[t]  = fa[i*HID + t] + b1[t];
    sW2[t] = W2r[t];
  }
  for (int t = threadIdx.x; t < 4*HID; t += 256) sG[t] = W1g[t];
#endif
  __syncthreads();
  float g0 = fabsf(boxes[i*4+0]-boxes[j*4+0]);
  float g1 = fabsf(boxes[i*4+1]-boxes[j*4+1]);
  float g2 = fabsf(boxes[i*4+2]-boxes[j*4+2]);
  float g3 = fabsf(boxes[i*4+3]-boxes[j*4+3]);
  float acc = 0.f;
  for (int f = 0; f < HID; ++f) {
    float v = sA[f] + fbT[f*N_NODES + j]
            + g0*sG[f] + g1*sG[HID+f] + g2*sG[2*HID+f] + g3*sG[3*HID+f];
    acc = fmaf(sW2[f], fmaxf(v, 0.f), acc);
  }
  out[(size_t)i*OUTC + j] = acc + b2r[0];
}

// ---------- top-K per row (stable on ties like jax top_k) ----------
__global__ void k_topk(const float* __restrict__ out, int* __restrict__ idx) {
  int i = blockIdx.x*blockDim.x + threadIdx.x;
  if (i >= N_NODES) return;
  float v[TOPK]; int ix[TOPK];
  #pragma unroll
  for (int k = 0; k < TOPK; ++k) { v[k] = -3.0e38f; ix[k] = 0; }
  const float* row = out + (size_t)i*OUTC;
  for (int j = 0; j < N_NODES; ++j) {
    if (j == i) continue;                   // diagonal excluded (-1e6 in reference)
    float r = row[j];
    if (r > v[TOPK-1]) {
      int p = TOPK-1;
      while (p > 0 && r > v[p-1]) { v[p] = v[p-1]; ix[p] = ix[p-1]; --p; }
      v[p] = r; ix[p] = j;
    }
  }
  for (int k = 0; k < TOPK; ++k) idx[i*TOPK + k] = ix[k];
}

// ---------- GAT pieces ----------
__global__ void k_alpha1(const float* __restrict__ h, const float* __restrict__ asrc,
                         const float* __restrict__ adst, float* __restrict__ as_,
                         float* __restrict__ ad_) {
  int t = blockIdx.x*blockDim.x + threadIdx.x;
  if (t >= N_NODES*HEADS) return;
  int n = t / HEADS, hd = t % HEADS;
  const float* hp = h + (size_t)n*(HEADS*HID) + hd*HID;
  const float* sp = asrc + hd*HID;
  const float* dp = adst + hd*HID;
  float s = 0.f, d = 0.f;
  for (int f = 0; f < HID; ++f) { s = fmaf(hp[f], sp[f], s); d = fmaf(hp[f], dp[f], d); }
  as_[t] = s; ad_[t] = d;
}

__device__ __forceinline__ void edge_decode(int eid, const int* __restrict__ tk,
                                            int& src, int& dst) {
  src = eid / SLOTS;
  int slot = eid % SLOTS;
  dst = (slot < TOPK) ? tk[src*TOPK + slot] : src;
}

__global__ void k_edge_max(const float* __restrict__ as_, const float* __restrict__ ad_,
                           const int* __restrict__ tk, unsigned* __restrict__ mkey, int heads) {
  int t = blockIdx.x*blockDim.x + threadIdx.x;
  if (t >= NEDGES*heads) return;
  int eid = t / heads, hd = t % heads;
  int src, dst; edge_decode(eid, tk, src, dst);
  float e = as_[src*heads + hd] + ad_[dst*heads + hd];
  e = (e >= 0.f) ? e : NEG_SLOPE*e;
  atomicMax(&mkey[dst*heads + hd], fkey(e));
}

__global__ void k_edge_exp(const float* __restrict__ as_, const float* __restrict__ ad_,
                           const int* __restrict__ tk, const unsigned* __restrict__ mkey,
                           float* __restrict__ ex, float* __restrict__ denom, int heads) {
  int t = blockIdx.x*blockDim.x + threadIdx.x;
  if (t >= NEDGES*heads) return;
  int eid = t / heads, hd = t % heads;
  int src, dst; edge_decode(eid, tk, src, dst);
  float e = as_[src*heads + hd] + ad_[dst*heads + hd];
  e = (e >= 0.f) ? e : NEG_SLOPE*e;
  float m = funkey(mkey[dst*heads + hd]);
  float v = expf(e - m);
  ex[t] = v;
  atomicAdd(&denom[dst*heads + hd], v);
}

__global__ void k_edge_aggr(const float* __restrict__ hfeat, const int* __restrict__ tk,
                            const float* __restrict__ ex, const float* __restrict__ denom,
                            float* __restrict__ aggr, int heads, int F) {
  int eid = blockIdx.x;
  int CH = heads * F;
  int src, dst; edge_decode(eid, tk, src, dst);
  for (int ch = threadIdx.x; ch < CH; ch += blockDim.x) {
    int hd = ch / F;
    float attn = ex[eid*heads + hd] / denom[dst*heads + hd];
    atomicAdd(&aggr[(size_t)dst*CH + ch], hfeat[(size_t)src*CH + ch] * attn);
  }
}

// h1 = relu(aggr + bg1) in place; also build bf16 reg_in = [h1 | geom | 0-pad]
__global__ void k_h1_regbf(float* __restrict__ aggr, const float* __restrict__ bg1,
                           const float* __restrict__ geom, __bf16* __restrict__ regbf) {
  int t = blockIdx.x*blockDim.x + threadIdx.x;
  int n = t / XKP, c = t % XKP;
  if (n >= N_NODES) return;
  float v;
  if (c < HEADS*HID) {
    v = fmaxf(aggr[(size_t)n*(HEADS*HID) + c] + bg1[c], 0.f);
    aggr[(size_t)n*(HEADS*HID) + c] = v;
  } else if (c < XK) v = geom[n*4 + (c - HEADS*HID)];
  else v = 0.f;
  regbf[(size_t)n*XKP + c] = f2bf(v);
}

// GAT2 features h2 = h1 @ Wg2 (1024x2) and its alphas
__global__ void k_h2a2(const float* __restrict__ h1, const float* __restrict__ Wg2,
                       const float* __restrict__ asrc2, const float* __restrict__ adst2,
                       float* __restrict__ h2, float* __restrict__ as2, float* __restrict__ ad2) {
  int n = blockIdx.x*blockDim.x + threadIdx.x;
  if (n >= N_NODES) return;
  const float* hp = h1 + (size_t)n*(HEADS*HID);
  float a = 0.f, b = 0.f;
  for (int f = 0; f < HEADS*HID; ++f) {
    a = fmaf(hp[f], Wg2[f*2+0], a);
    b = fmaf(hp[f], Wg2[f*2+1], b);
  }
  h2[n*2+0] = a; h2[n*2+1] = b;
  as2[n] = a*asrc2[0] + b*asrc2[1];
  ad2[n] = a*adst2[0] + b*adst2[1];
}

// logits/softmax, regression head tail, box decode -> out cols 768..774
__global__ void k_final(const float* __restrict__ lacc, const float* __restrict__ bg2,
                        const float* __restrict__ hidden, const float* __restrict__ bb1,
                        const float* __restrict__ Wb2, const float* __restrict__ bb2,
                        const float* __restrict__ boxes, float* __restrict__ out) {
  int n = blockIdx.x*blockDim.x + threadIdx.x;
  if (n >= N_NODES) return;
  float l0 = lacc[n*2+0] + bg2[0];
  float l1 = lacc[n*2+1] + bg2[1];
  float mx = fmaxf(l0, l1);
  float e0 = expf(l0-mx), e1 = expf(l1-mx);
  float prob = e1 / (e0 + e1);
  float d0 = bb2[0], d1 = bb2[1], d2 = bb2[2], d3 = bb2[3];
  const float* hp = hidden + (size_t)n*(HID/2);
  for (int q = 0; q < HID/2; ++q) {
    float hv = fmaxf(hp[q] + bb1[q], 0.f);
    d0 = fmaf(hv, Wb2[q*4+0], d0);
    d1 = fmaf(hv, Wb2[q*4+1], d1);
    d2 = fmaf(hv, Wb2[q*4+2], d2);
    d3 = fmaf(hv, Wb2[q*4+3], d3);
  }
  float x0 = boxes[n*4+0], y0 = boxes[n*4+1], x1 = boxes[n*4+2], y1 = boxes[n*4+3];
  float pw = x1-x0, ph = y1-y0, pcx = x0 + 0.5f*pw, pcy = y0 + 0.5f*ph;
  float gcx = fmaf(d0, pw, pcx), gcy = fmaf(d1, ph, pcy);
  float gw = expf(d2)*pw, gh = expf(d3)*ph;
  float* o = out + (size_t)n*OUTC;
  o[768] = l0; o[769] = l1;
  o[770] = gcx - 0.5f*gw; o[771] = gcy - 0.5f*gh;
  o[772] = gcx + 0.5f*gw; o[773] = gcy + 0.5f*gh;
  o[774] = prob;
}

// ---------- host ----------
extern "C" void kernel_launch(void* const* d_in, const int* in_sizes, int n_in,
                              void* d_out, int out_size, void* d_ws, size_t ws_size,
                              hipStream_t stream) {
  (void)in_sizes; (void)n_in; (void)out_size; (void)ws_size;
  const float* feats = (const float*)d_in[0];
  const float* boxes = (const float*)d_in[1];
  const int*   img_h = (const int*)d_in[2];
  const int*   img_w = (const int*)d_in[3];
  const float* W1f1  = (const float*)d_in[4];
  const float* W1f2  = (const float*)d_in[5];
  const float* W1g   = (const float*)d_in[6];
  const float* b1    = (const float*)d_in[7];
  const float* W2r   = (const float*)d_in[8];
  const float* b2r   = (const float*)d_in[9];
  const float* Wg1   = (const float*)d_in[10];
  const float* asrc1 = (const float*)d_in[11];
  const float* adst1 = (const float*)d_in[12];
  const float* bg1   = (const float*)d_in[13];
  const float* Wg2   = (const float*)d_in[14];
  const float* asrc2 = (const float*)d_in[15];
  const float* adst2 = (const float*)d_in[16];
  const float* bg2   = (const float*)d_in[17];
  const float* Wb1   = (const float*)d_in[18];
  const float* bb1   = (const float*)d_in[19];
  const float* Wb2   = (const float*)d_in[20];
  const float* bb2   = (const float*)d_in[21];
  float* out = (float*)d_out;

  char* p = (char*)d_ws;
  auto alloc = [&](size_t bytes) -> char* {
    char* r = p; p += (bytes + 255) & ~(size_t)255; return r;
  };
  __bf16* xbf    = (__bf16*)alloc((size_t)N_NODES*XKP*2);
  __bf16* w1f1b  = (__bf16*)alloc((size_t)C_FEAT*HID*2);
  __bf16* w1f2b  = (__bf16*)alloc((size_t)C_FEAT*HID*2);
  __bf16* wg1b   = (__bf16*)alloc((size_t)XKP*1024*2);
  __bf16* wb1b   = (__bf16*)alloc((size_t)XKP*128*2);
  __bf16* regbf  = (__bf16*)alloc((size_t)N_NODES*XKP*2);
  float*  geom   = (float*)alloc((size_t)N_NODES*4*4);
  float*  fa     = (float*)alloc((size_t)N_NODES*HID*4);
  float*  fbT    = (float*)alloc((size_t)HID*N_NODES*4);
  float*  hbuf   = (float*)alloc((size_t)N_NODES*1024*4);
  float*  as1    = (float*)alloc((size_t)N_NODES*HEADS*4);
  float*  ad1    = (float*)alloc((size_t)N_NODES*HEADS*4);
  int*    tk     = (int*)alloc((size_t)N_NODES*TOPK*4);
  unsigned* mk1  = (unsigned*)alloc((size_t)N_NODES*HEADS*4);
  float*  den1   = (float*)alloc((size_t)N_NODES*HEADS*4);
  float*  ex1    = (float*)alloc((size_t)NEDGES*HEADS*4);
  float*  aggr1  = (float*)alloc((size_t)N_NODES*1024*4);
  float*  h2     = (float*)alloc((size_t)N_NODES*2*4);
  float*  as2    = (float*)alloc((size_t)N_NODES*4);
  float*  ad2    = (float*)alloc((size_t)N_NODES*4);
  unsigned* mk2  = (unsigned*)alloc((size_t)N_NODES*4);
  float*  den2   = (float*)alloc((size_t)N_NODES*4);
  float*  ex2    = (float*)alloc((size_t)NEDGES*4);
  float*  lacc   = (float*)alloc((size_t)N_NODES*2*4);
  float*  hidden = (float*)alloc((size_t)N_NODES*128*4);

  const int T = 256;
  // stage 0: staging
  k_build_x<<<(N_NODES*XKP + T-1)/T, T, 0, stream>>>(feats, boxes, img_h, img_w, geom, xbf);
  k_castpad<<<(C_FEAT*HID + T-1)/T, T, 0, stream>>>(W1f1, w1f1b, C_FEAT, HID, C_FEAT);
  k_castpad<<<(C_FEAT*HID + T-1)/T, T, 0, stream>>>(W1f2, w1f2b, C_FEAT, HID, C_FEAT);
  k_castpad<<<(XKP*1024 + T-1)/T, T, 0, stream>>>(Wg1, wg1b, XK, 1024, XKP);
  k_castpad<<<(XKP*128 + T-1)/T, T, 0, stream>>>(Wb1, wb1b, XK, 128, XKP);
  // stage 1: fa = feats@W1_f1 ; fbT = (feats@W1_f2)^T  (WMMA, K=1024 slice of xbf)
  k_wmma_gemm<<<dim3(N_NODES/16, HID/64), 32, 0, stream>>>(xbf, w1f1b, fa, C_FEAT, XKP, HID, HID, 0);
  k_wmma_gemm<<<dim3(N_NODES/16, HID/64), 32, 0, stream>>>(xbf, w1f2b, fbT, C_FEAT, XKP, HID, N_NODES, 1);
  // stage 2: fused rel -> out cols [0,768)
  k_rel<<<dim3(3, N_NODES), 256, 0, stream>>>(fa, fbT, boxes, W1g, b1, W2r, b2r, out);
  // stage 3: top-K edges
  k_topk<<<(N_NODES + T-1)/T, T, 0, stream>>>(out, tk);
  // stage 4: GAT1 h = x @ Wg1 (WMMA, K=1056)
  k_wmma_gemm<<<dim3(N_NODES/16, 1024/64), 32, 0, stream>>>(xbf, wg1b, hbuf, XKP, XKP, 1024, 1024, 0);
  k_alpha1<<<(N_NODES*HEADS + T-1)/T, T, 0, stream>>>(hbuf, asrc1, adst1, as1, ad1);
  // zero accumulators (every call; graph-replay safe)
  k_zero<<<(N_NODES*HEADS + T-1)/T, T, 0, stream>>>((float*)mk1, N_NODES*HEADS);
  k_zero<<<(N_NODES*HEADS + T-1)/T, T, 0, stream>>>(den1, N_NODES*HEADS);
  k_zero<<<(N_NODES*1024 + T-1)/T, T, 0, stream>>>(aggr1, N_NODES*1024);
  k_zero<<<(N_NODES + T-1)/T, T, 0, stream>>>((float*)mk2, N_NODES);
  k_zero<<<(N_NODES + T-1)/T, T, 0, stream>>>(den2, N_NODES);
  k_zero<<<(N_NODES*2 + T-1)/T, T, 0, stream>>>(lacc, N_NODES*2);
  // stage 5: GAT1 segment softmax + aggregation
  k_edge_max <<<(NEDGES*HEADS + T-1)/T, T, 0, stream>>>(as1, ad1, tk, mk1, HEADS);
  k_edge_exp <<<(NEDGES*HEADS + T-1)/T, T, 0, stream>>>(as1, ad1, tk, mk1, ex1, den1, HEADS);
  k_edge_aggr<<<NEDGES, 256, 0, stream>>>(hbuf, tk, ex1, den1, aggr1, HEADS, HID);
  // stage 6: h1 = relu(aggr+bg1), build reg_in bf16
  k_h1_regbf<<<(N_NODES*XKP + T-1)/T, T, 0, stream>>>(aggr1, bg1, geom, regbf);
  // stage 7: GAT2
  k_h2a2<<<(N_NODES + T-1)/T, T, 0, stream>>>(aggr1, Wg2, asrc2, adst2, h2, as2, ad2);
  k_edge_max <<<(NEDGES + T-1)/T, T, 0, stream>>>(as2, ad2, tk, mk2, 1);
  k_edge_exp <<<(NEDGES + T-1)/T, T, 0, stream>>>(as2, ad2, tk, mk2, ex2, den2, 1);
  k_edge_aggr<<<NEDGES, 32, 0, stream>>>(h2, tk, ex2, den2, lacc, 1, 2);
  // stage 8: regression hidden = reg_in @ Wb1 (WMMA, K=1056)
  k_wmma_gemm<<<dim3(N_NODES/16, 128/64), 32, 0, stream>>>(regbf, wb1b, hidden, XKP, XKP, 128, 128, 0);
  // stage 9: logits/softmax/delta/box decode -> out cols [768,775)
  k_final<<<(N_NODES + T-1)/T, T, 0, stream>>>(lacc, bg2, hidden, bb1, Wb2, bb2, boxes, out);
}
